// GroupedExpert_6768868459218
// MI455X (gfx1250) — matrix operand order repaired and checked by reading, not compiled, not run on hardware
//
#include <hip/hip_runtime.h>
#include <hip/hip_bf16.h>

// ---------------------------------------------------------------------------
// Grouped SwiGLU expert MLP for gfx1250 (MI455X), bf16 WMMA + async-LDS path.
//   h = silu(x @ w1[e]) * (x @ w3[e]);  out = h @ w2[e]   (ragged over experts)
// E=8, D=2048, F=512, T=32768.
// ---------------------------------------------------------------------------

#define E_ 8
#define D_ 2048
#define F_ 512
#define T_ 32768
#define BM 128
#define BK 32
#define MAX_TILES (T_ / BM + E_)   // 264 upper bound on ragged row tiles

typedef __attribute__((ext_vector_type(16))) __bf16 v16bf;
typedef __attribute__((ext_vector_type(8)))  float  v8f;
typedef __attribute__((ext_vector_type(4)))  int    v4i;
typedef unsigned short u16;
typedef unsigned int   u32;

struct B32x2 { uint4 lo; uint4 hi; };   // 32 bytes -> one v16bf fragment

static __device__ __forceinline__ u16 f2bf(float f) {
    union { float f; u32 u; } v; v.f = f;
    u32 r = v.u + 0x7FFFu + ((v.u >> 16) & 1u);   // round-to-nearest-even
    return (u16)(r >> 16);
}
static __device__ __forceinline__ u32 pack2bf(float a, float b) {
    return (u32)f2bf(a) | ((u32)f2bf(b) << 16);
}
static __device__ __forceinline__ v16bf load_frag(const u16* p0, const u16* p1) {
    B32x2 f;
    f.lo = *(const uint4*)p0;   // ds_load_b128
    f.hi = *(const uint4*)p1;   // ds_load_b128
    return __builtin_bit_cast(v16bf, f);
}

// ---- CDNA5 async global->LDS copy (16B per lane), ASYNCcnt-tracked -------
static __device__ __forceinline__ void async_cp16(const u16* gsrc, u16* ldst) {
#if __has_builtin(__builtin_amdgcn_global_load_async_to_lds_b128)
    __builtin_amdgcn_global_load_async_to_lds_b128(
        (__attribute__((address_space(1))) v4i*)(gsrc),
        (__attribute__((address_space(3))) v4i*)(ldst), 0, 0);
#else
    unsigned laddr = (unsigned)(__UINTPTR_TYPE__)(__attribute__((address_space(3))) void*)(ldst);
    asm volatile("global_load_async_to_lds_b128 %0, %1, off"
                 :: "v"(laddr), "v"(gsrc) : "memory");
#endif
}
static __device__ __forceinline__ void wait_async0() {
#if __has_builtin(__builtin_amdgcn_s_wait_asynccnt)
    __builtin_amdgcn_s_wait_asynccnt(0);
#else
    asm volatile("s_wait_asynccnt 0x0" ::: "memory");
#endif
}

// Map linear row-tile id -> (expert, tile row start, #valid rows). Uniform per block.
static __device__ __forceinline__ bool map_tile_e(const int* __restrict__ counts,
                                                  int tile, int& e_out, int& m0, int& rows_valid) {
    int e = -1, lt = 0, row_off = 0, cnt = 0, tacc = 0, racc = 0;
    #pragma unroll
    for (int i = 0; i < E_; ++i) {
        int c  = counts[i];
        int te = (c + BM - 1) >> 7;
        if (e < 0 && tile < tacc + te) { e = i; lt = tile - tacc; row_off = racc; cnt = c; }
        tacc += te; racc += c;
    }
    if (e < 0) return false;
    e_out = e;
    m0 = row_off + lt * BM;
    rows_valid = cnt - lt * BM; if (rows_valid > BM) rows_valid = BM;
    return true;
}

// ---------------------------------------------------------------------------
// Tiled f32 -> bf16 transpose: in[b][r][c] (f32) -> out[b][c][r] (bf16)
// ---------------------------------------------------------------------------
__global__ __launch_bounds__(256)
void transpose_bf16_k(const float* __restrict__ in, u16* __restrict__ out,
                      int R, int C, long long outBatchStride) {
    __shared__ float tile[32][33];
    const int b  = blockIdx.z;
    const float* ib = in + (size_t)b * R * C;
    u16* ob = out + (size_t)b * outBatchStride;
    const int c0 = blockIdx.x * 32, r0 = blockIdx.y * 32;
    const int tx = threadIdx.x, ty = threadIdx.y;      // (32, 8)
    #pragma unroll
    for (int j = 0; j < 32; j += 8)
        tile[ty + j][tx] = ib[(size_t)(r0 + ty + j) * C + (c0 + tx)];
    __syncthreads();
    #pragma unroll
    for (int j = 0; j < 32; j += 8)
        ob[(size_t)(c0 + ty + j) * R + (r0 + tx)] = f2bf(tile[tx][ty + j]);
}

// ---------------------------------------------------------------------------
// GEMM1 fused: x(f32)[rows,D] @ {w1,w3}(bf16,n-major) -> h(bf16)[rows,F]
// Block: 256 thr (8 wave32s, 4x2). Tile 128x128, K-step 32, double-buffered:
// B tiles via async global->LDS; A tile converted f32->bf16 through VGPRs.
// ---------------------------------------------------------------------------
__global__ __launch_bounds__(256)
void gemm1_swiglu_k(const float* __restrict__ x, const int* __restrict__ counts,
                    const u16* __restrict__ w13t, u16* __restrict__ hb) {
    int e, m0, rows_valid;
    if (!map_tile_e(counts, (int)blockIdx.x, e, m0, rows_valid)) return;
    const int n0 = blockIdx.y * 128;

    __shared__ alignas(16) u16 sA [2][BM * BK];
    __shared__ alignas(16) u16 sB1[2][128 * BK];
    __shared__ alignas(16) u16 sB3[2][128 * BK];

    const int lane = threadIdx.x & 31;
    const int wv   = threadIdx.x >> 5;
    const int wm   = wv & 3;            // 4 waves along M (32 rows each)
    const int wn   = wv >> 2;           // 2 waves along N (64 cols each)
    const int arow = lane & 15;
    const int asel = (lane >> 4) * 16;  // A frag K-chunk select
    const int bsel = (lane >> 4) * 16;  // B frag K-chunk select

    v8f acc1[2][4], acc3[2][4];
    const v8f vz = {0.f,0.f,0.f,0.f,0.f,0.f,0.f,0.f};
    #pragma unroll
    for (int i = 0; i < 2; ++i)
        #pragma unroll
        for (int j = 0; j < 4; ++j) { acc1[i][j] = vz; acc3[i][j] = vz; }

    const u16* w1b = w13t + ((size_t)(e * 2 + 0) * F_) * D_;
    const u16* w3b = w13t + ((size_t)(e * 2 + 1) * F_) * D_;

    auto stage_B = [&](int k0, int buf) {   // async global -> LDS, 4x b128/thread
        #pragma unroll
        for (int pass = 0; pass < 2; ++pass) {
            const int idx  = (int)threadIdx.x + pass * 256;
            const int n    = idx >> 2;
            const int part = (idx & 3) * 8;
            const size_t off = (size_t)(n0 + n) * D_ + k0 + part;
            async_cp16(w1b + off, &sB1[buf][n * BK + part]);
            async_cp16(w3b + off, &sB3[buf][n * BK + part]);
        }
    };
    auto stage_A = [&](int k0, int buf) {   // f32 -> bf16 convert through VGPRs
        const int t   = (int)threadIdx.x;
        const int row = t >> 1;
        const int kq  = (t & 1) * 16;
        const int rc  = row < rows_valid ? row : rows_valid - 1;  // clamp (masked at store)
        const float4* src = (const float4*)(x + (size_t)(m0 + rc) * D_ + k0 + kq);
        float4 f0 = src[0], f1 = src[1], f2 = src[2], f3 = src[3];
        uint4 p0 = make_uint4(pack2bf(f0.x,f0.y), pack2bf(f0.z,f0.w),
                              pack2bf(f1.x,f1.y), pack2bf(f1.z,f1.w));
        uint4 p1 = make_uint4(pack2bf(f2.x,f2.y), pack2bf(f2.z,f2.w),
                              pack2bf(f3.x,f3.y), pack2bf(f3.z,f3.w));
        *(uint4*)&sA[buf][row * BK + kq]     = p0;
        *(uint4*)&sA[buf][row * BK + kq + 8] = p1;
    };

    // Prologue: fill buffer 0.
    stage_B(0, 0);
    stage_A(0, 0);
    wait_async0();
    __syncthreads();

    for (int k0 = 0; k0 < D_; k0 += BK) {
        const int cur = (k0 >> 5) & 1;
        if (k0 + BK < D_) {               // stage next slab into the other buffer
            stage_B(k0 + BK, cur ^ 1);
            stage_A(k0 + BK, cur ^ 1);
        }
        // ---- compute on current buffer
        v16bf fa[2];
        #pragma unroll
        for (int mi = 0; mi < 2; ++mi) {
            const int r = wm * 32 + mi * 16 + arow;
            fa[mi] = load_frag(&sA[cur][r * BK + asel], &sA[cur][r * BK + asel + 8]);
        }
        #pragma unroll
        for (int ni = 0; ni < 4; ++ni) {
            const int c = wn * 64 + ni * 16 + arow;
            v16bf fb1 = load_frag(&sB1[cur][c * BK + bsel], &sB1[cur][c * BK + bsel + 8]);
            v16bf fb3 = load_frag(&sB3[cur][c * BK + bsel], &sB3[cur][c * BK + bsel + 8]);
            #pragma unroll
            for (int mi = 0; mi < 2; ++mi) {
                acc1[mi][ni] = __builtin_amdgcn_wmma_f32_16x16x32_bf16(
                    false, fa[mi], false, fb1, (short)0, acc1[mi][ni], false, false);
                acc3[mi][ni] = __builtin_amdgcn_wmma_f32_16x16x32_bf16(
                    false, fa[mi], false, fb3, (short)0, acc3[mi][ni], false, false);
            }
        }
        wait_async0();      // async writes to next buffer landed
        __syncthreads();    // + dscnt: publish A stores, retire reads of cur
    }

    // ---- epilogue: h = silu(x1) * x3, bf16 store (row-guarded)
    #pragma unroll
    for (int mi = 0; mi < 2; ++mi) {
        const int mb = wm * 32 + mi * 16 + (lane >> 4) * 8;
        #pragma unroll
        for (int ni = 0; ni < 4; ++ni) {
            const int nn = n0 + wn * 64 + ni * 16 + (lane & 15);
            #pragma unroll
            for (int r = 0; r < 8; ++r) {
                const int m = mb + r;
                if (m < rows_valid) {
                    float a = acc1[mi][ni][r];
                    float h = (a / (1.0f + __expf(-a))) * acc3[mi][ni][r];
                    hb[(size_t)(m0 + m) * F_ + nn] = f2bf(h);
                }
            }
        }
    }
}

// ---------------------------------------------------------------------------
// GEMM2: h(bf16)[rows,F] @ w2(bf16,n-major [E][D][F]) -> out(f32)[rows,D]
// Both A and B tiles move via async global->LDS; double-buffered.
// ---------------------------------------------------------------------------
__global__ __launch_bounds__(256)
void gemm2_k(const u16* __restrict__ hb, const int* __restrict__ counts,
             const u16* __restrict__ w2t, float* __restrict__ out) {
    int e, m0, rows_valid;
    if (!map_tile_e(counts, (int)blockIdx.x, e, m0, rows_valid)) return;
    const int n0 = blockIdx.y * 128;

    __shared__ alignas(16) u16 sA[2][BM * BK];
    __shared__ alignas(16) u16 sB[2][128 * BK];

    const int lane = threadIdx.x & 31;
    const int wv   = threadIdx.x >> 5;
    const int wm   = wv & 3;
    const int wn   = wv >> 2;
    const int arow = lane & 15;
    const int asel = (lane >> 4) * 16;
    const int bsel = (lane >> 4) * 16;

    v8f acc[2][4];
    const v8f vz = {0.f,0.f,0.f,0.f,0.f,0.f,0.f,0.f};
    #pragma unroll
    for (int i = 0; i < 2; ++i)
        #pragma unroll
        for (int j = 0; j < 4; ++j) acc[i][j] = vz;

    const u16* wb = w2t + (size_t)e * D_ * F_;

    auto stage = [&](int k0, int buf) {   // async global -> LDS for A and B
        #pragma unroll
        for (int pass = 0; pass < 2; ++pass) {
            const int idx  = (int)threadIdx.x + pass * 256;
            const int row  = idx >> 2;
            const int part = (idx & 3) * 8;
            const int rc   = row < rows_valid ? row : rows_valid - 1;
            async_cp16(hb + (size_t)(m0 + rc) * F_ + k0 + part, &sA[buf][row * BK + part]);
            async_cp16(wb + (size_t)(n0 + row) * F_ + k0 + part, &sB[buf][row * BK + part]);
        }
    };

    stage(0, 0);
    wait_async0();
    __syncthreads();

    for (int k0 = 0; k0 < F_; k0 += BK) {
        const int cur = (k0 >> 5) & 1;
        if (k0 + BK < F_) stage(k0 + BK, cur ^ 1);

        v16bf fa[2];
        #pragma unroll
        for (int mi = 0; mi < 2; ++mi) {
            const int r = wm * 32 + mi * 16 + arow;
            fa[mi] = load_frag(&sA[cur][r * BK + asel], &sA[cur][r * BK + asel + 8]);
        }
        #pragma unroll
        for (int ni = 0; ni < 4; ++ni) {
            const int c = wn * 64 + ni * 16 + arow;
            v16bf fb = load_frag(&sB[cur][c * BK + bsel], &sB[cur][c * BK + bsel + 8]);
            #pragma unroll
            for (int mi = 0; mi < 2; ++mi) {
                acc[mi][ni] = __builtin_amdgcn_wmma_f32_16x16x32_bf16(
                    false, fa[mi], false, fb, (short)0, acc[mi][ni], false, false);
            }
        }
        wait_async0();
        __syncthreads();
    }

    #pragma unroll
    for (int mi = 0; mi < 2; ++mi) {
        const int mb = wm * 32 + mi * 16 + (lane >> 4) * 8;
        #pragma unroll
        for (int ni = 0; ni < 4; ++ni) {
            const int nn = n0 + wn * 64 + ni * 16 + (lane & 15);
            #pragma unroll
            for (int r = 0; r < 8; ++r) {
                const int m = mb + r;
                if (m < rows_valid)
                    out[(size_t)(m0 + m) * D_ + nn] = acc[mi][ni][r];
            }
        }
    }
}

// ---------------------------------------------------------------------------
// Host launch. Inputs (dict order): x, num_tokens_per_expert, w1, w2, w3.
// Workspace: w13t (2*E*F*D bf16) | w2t (E*D*F bf16) | hb (T*F bf16)  = 80 MB.
// ---------------------------------------------------------------------------
extern "C" void kernel_launch(void* const* d_in, const int* in_sizes, int n_in,
                              void* d_out, int out_size, void* d_ws, size_t ws_size,
                              hipStream_t stream) {
    (void)in_sizes; (void)n_in; (void)out_size; (void)ws_size;
    const float* x      = (const float*)d_in[0];
    const int*   counts = (const int*)  d_in[1];
    const float* w1     = (const float*)d_in[2];
    const float* w2     = (const float*)d_in[3];
    const float* w3     = (const float*)d_in[4];
    float* out = (float*)d_out;

    u16* wsW13 = (u16*)d_ws;                                   // [E][2][F][D] bf16
    u16* wsW2  = wsW13 + (size_t)2 * E_ * F_ * D_;             // [E][D][F] bf16
    u16* wsH   = wsW2  + (size_t)E_ * D_ * F_;                 // [T][F] bf16

    // Convert + transpose weights to bf16 n-major (k contiguous).
    transpose_bf16_k<<<dim3(F_ / 32, D_ / 32, E_), dim3(32, 8), 0, stream>>>(
        w1, wsW13,            D_, F_, (long long)2 * F_ * D_);
    transpose_bf16_k<<<dim3(F_ / 32, D_ / 32, E_), dim3(32, 8), 0, stream>>>(
        w3, wsW13 + (size_t)F_ * D_, D_, F_, (long long)2 * F_ * D_);
    transpose_bf16_k<<<dim3(D_ / 32, F_ / 32, E_), dim3(32, 8), 0, stream>>>(
        w2, wsW2,             F_, D_, (long long)D_ * F_);

    // Fused gate GEMM: h = silu(x@w1) * (x@w3)
    gemm1_swiglu_k<<<dim3(MAX_TILES, F_ / 128), dim3(256), 0, stream>>>(
        x, counts, wsW13, wsH);

    // Down projection: out = h @ w2
    gemm2_k<<<dim3(MAX_TILES, D_ / 128), dim3(256), 0, stream>>>(
        wsH, counts, wsW2, out);
}
